// PredictiveHierarchyEngine_72902774882817
// MI455X (gfx1250) — compile-verified, matrix operand order checked by reading, not run on hardware
//
#include <hip/hip_runtime.h>
#include <hip/hip_bf16.h>
#include <cstdint>

// ---------------------------------------------------------------------------
// Types for WMMA bf16 (gfx1250: V_WMMA_F32_16X16X32_BF16)
// ---------------------------------------------------------------------------
typedef __attribute__((ext_vector_type(16))) __bf16        bfrag_t;
typedef __attribute__((ext_vector_type(8)))  float         v8f;
typedef __attribute__((ext_vector_type(4)))  unsigned int  v4u;
typedef __attribute__((ext_vector_type(8)))  unsigned int  v8u;

union FragCvt { v8u u; bfrag_t bf; };

__device__ __forceinline__ v8f wmma_bf16(bfrag_t a, bfrag_t b, v8f c) {
  // (neg_a, A, neg_b, B, c_mod, C, reuse_a, reuse_b)
  return __builtin_amdgcn_wmma_f32_16x16x32_bf16(false, a, false, b, (short)0, c,
                                                 false, false);
}

__device__ __forceinline__ unsigned short f32_to_bf16(float f) {
  unsigned int u = __float_as_uint(f);
  unsigned int r = u + 0x7FFFu + ((u >> 16) & 1u);   // round-to-nearest-even
  return (unsigned short)(r >> 16);
}

// Load a 16x32 bf16 operand fragment from a row-major [rows][ldk] bf16 array.
// ISA layout: lane<16 -> M=lane, K in {k0..k0+7, k0+16..k0+23};
//             lane>=16 -> same M, K ranges shifted by +8.
// ldk == 0 gives a broadcast row (all lanes read the same slice).
__device__ __forceinline__ bfrag_t load_frag(const unsigned short* __restrict__ base,
                                             int row0, int ldk, int k0) {
  const int lane  = threadIdx.x & 31;
  const int r     = lane & 15;
  const int khalf = (lane >> 4) & 1;
  const unsigned short* p = base + (size_t)(row0 + r) * ldk + k0 + khalf * 8;
  v4u lo = *(const v4u*)(p);
  v4u hi = *(const v4u*)(p + 16);
  FragCvt f;
  f.u[0] = lo[0]; f.u[1] = lo[1]; f.u[2] = lo[2]; f.u[3] = lo[3];
  f.u[4] = hi[0]; f.u[5] = hi[1]; f.u[6] = hi[2]; f.u[7] = hi[3];
  return f.bf;
}

// ---------------------------------------------------------------------------
// Problem constants
// ---------------------------------------------------------------------------
#define L_LV   4
#define CPL    1024
#define IN_D   512
#define HID    1024
#define OUT_D  512
#define MLP_D  128
#define COMB_D (IN_D + HID)     // 1536
#define G3     (3 * HID)        // 3072

// ---------------------------------------------------------------------------
// Conversion kernels (fp32 -> bf16, with transposes where the GEMM needs
// the B operand laid out as [N][K] row-major)
// ---------------------------------------------------------------------------
__global__ void k_cvt_bf16(const float* __restrict__ src,
                           unsigned short* __restrict__ dst, int n) {
  int i = blockIdx.x * 256 + threadIdx.x;
  if (i < n) dst[i] = f32_to_bf16(src[i]);
}

// [L][1536][128] -> [L][128][1536]
__global__ void k_transpose_W1(const float* __restrict__ src,
                               unsigned short* __restrict__ dst) {
  int i = blockIdx.x * 256 + threadIdx.x;
  if (i >= L_LV * MLP_D * COMB_D) return;
  int l = i / (MLP_D * COMB_D);
  int r = i % (MLP_D * COMB_D);
  int m = r / COMB_D;
  int k = r % COMB_D;
  dst[i] = f32_to_bf16(src[(size_t)l * COMB_D * MLP_D + (size_t)k * MLP_D + m]);
}

// [L][128][512] -> [L][512][128]
__global__ void k_transpose_W2(const float* __restrict__ src,
                               unsigned short* __restrict__ dst) {
  int i = blockIdx.x * 256 + threadIdx.x;
  if (i >= L_LV * OUT_D * MLP_D) return;
  int l = i / (OUT_D * MLP_D);
  int r = i % (OUT_D * MLP_D);
  int o = r / MLP_D;
  int m = r % MLP_D;
  dst[i] = f32_to_bf16(src[(size_t)l * MLP_D * OUT_D + (size_t)m * OUT_D + o]);
}

// gru_Wih [L][3072][513]: first 512 cols -> bf16 [L*3072][512], col 512 -> f32
__global__ void k_cvt_wih(const float* __restrict__ src,
                          unsigned short* __restrict__ dst,
                          float* __restrict__ lastcol) {
  int i = blockIdx.x * 256 + threadIdx.x;
  if (i >= L_LV * G3 * OUT_D) return;
  int g = i >> 9;        // combined l*3072+g
  int k = i & 511;
  dst[i] = f32_to_bf16(src[(size_t)g * (OUT_D + 1) + k]);
  if (k == 0) lastcol[g] = src[(size_t)g * (OUT_D + 1) + OUT_D];
}

// ---------------------------------------------------------------------------
// Phase 1: out = MLP_a(comb) - MLP_g(comb), tension = mean(out^2)
// One WG per (level, 64-cell tile). 8 waves; wave w -> GEMM1 col tile w,
// GEMM2 cols w*64..w*64+63. A fragments loaded directly from global
// (x part is a broadcast row: ldk = 0). Barriers only around the Z transpose.
// ---------------------------------------------------------------------------
__global__ __launch_bounds__(256)
void k_mlp(const unsigned short* __restrict__ x_bf,
           const unsigned short* __restrict__ h_bf,
           const unsigned short* __restrict__ WA1t, const float* __restrict__ ba1,
           const unsigned short* __restrict__ WA2t, const float* __restrict__ ba2,
           const unsigned short* __restrict__ WG1t, const float* __restrict__ bg1,
           const unsigned short* __restrict__ WG2t, const float* __restrict__ bg2,
           unsigned short* __restrict__ out_bf, float* __restrict__ tension) {
  __shared__ __align__(16) unsigned short Zs[64 * 128];   // z tile [64 cells][128]
  __shared__ float tens[64];

  const int l    = blockIdx.x >> 4;
  const int c0   = (blockIdx.x & 15) * 64;
  const int w    = threadIdx.x >> 5;
  const int lane = threadIdx.x & 31;
  const int hi8  = (lane >> 4) * 8;

  const unsigned short* Ah = h_bf + (size_t)(l * CPL + c0) * HID;   // h part of comb

  v8f acc2[4][4];
  for (int mt = 0; mt < 4; ++mt)
    for (int nt = 0; nt < 4; ++nt) acc2[mt][nt] = (v8f){};

  for (int pass = 0; pass < 2; ++pass) {
    const unsigned short* W1t = pass ? WG1t : WA1t;
    const unsigned short* W2t = pass ? WG2t : WA2t;
    const float*          b1  = pass ? bg1  : ba1;
    const float           sgn = pass ? -1.0f : 1.0f;

    // ---- GEMM1: z[64][128] = comb[64][1536] @ W1 (barrier-free) ----
    v8f acc1[4];
    for (int mt = 0; mt < 4; ++mt) acc1[mt] = (v8f){};

    const unsigned short* W1L = W1t + (size_t)l * MLP_D * COMB_D;
    // x region: same row for every cell -> broadcast fragment (ldk = 0)
    for (int kt = 0; kt < IN_D; kt += 32) {
      bfrag_t bf = load_frag(W1L, w * 16, COMB_D, kt);
      bfrag_t ax = load_frag(x_bf, 0, 0, kt);
      for (int mt = 0; mt < 4; ++mt)
        acc1[mt] = wmma_bf16(ax, bf, acc1[mt]);
    }
    // h region
    for (int kt = 0; kt < HID; kt += 32) {
      bfrag_t bf = load_frag(W1L, w * 16, COMB_D, IN_D + kt);
      for (int mt = 0; mt < 4; ++mt) {
        bfrag_t af = load_frag(Ah, mt * 16, HID, kt);
        acc1[mt] = wmma_bf16(af, bf, acc1[mt]);
      }
    }

    // ---- relu + bias, stash (signed) bf16 z into LDS ----
    __syncthreads();                       // previous pass's GEMM2 reads done
    {
      const int colz = w * 16 + (lane & 15);
      const float bb = b1[l * MLP_D + colz];
      for (int mt = 0; mt < 4; ++mt)
        for (int i = 0; i < 8; ++i) {
          float z = acc1[mt][i] + bb;
          z = z > 0.0f ? z : 0.0f;
          Zs[(mt * 16 + hi8 + i) * MLP_D + colz] = f32_to_bf16(sgn * z);
        }
    }
    __syncthreads();

    // ---- GEMM2: out[64][512] += z[64][128] @ W2 (barrier-free) ----
    for (int kt = 0; kt < MLP_D; kt += 32) {
      bfrag_t bfr[4];
      for (int nt = 0; nt < 4; ++nt)
        bfr[nt] = load_frag(W2t + (size_t)l * OUT_D * MLP_D,
                            w * 64 + nt * 16, MLP_D, kt);
      for (int mt = 0; mt < 4; ++mt) {
        bfrag_t af = load_frag(Zs, mt * 16, MLP_D, kt);
        for (int nt = 0; nt < 4; ++nt)
          acc2[mt][nt] = wmma_bf16(af, bfr[nt], acc2[mt][nt]);
      }
    }
  }

  // ---- epilogue: bias diff, store bf16 out, row-wise tension ----
  __syncthreads();
  if (threadIdx.x < 64) tens[threadIdx.x] = 0.0f;
  __syncthreads();
  {
    const int coln = lane & 15;
    for (int nt = 0; nt < 4; ++nt) {
      const int col = w * 64 + nt * 16 + coln;
      const float b2d = ba2[l * OUT_D + col] - bg2[l * OUT_D + col];
      for (int mt = 0; mt < 4; ++mt)
        for (int i = 0; i < 8; ++i) {
          int m = mt * 16 + hi8 + i;
          float v = acc2[mt][nt][i] + b2d;
          out_bf[(size_t)(l * CPL + c0 + m) * OUT_D + col] = f32_to_bf16(v);
          atomicAdd(&tens[m], v * v);
        }
    }
  }
  __syncthreads();
  if (threadIdx.x < 64)
    tension[l * CPL + c0 + threadIdx.x] = tens[threadIdx.x] * (1.0f / OUT_D);
}

// ---------------------------------------------------------------------------
// Phase 2: GRU. One WG per (level, 32-cell tile, 128-col tile); wave w owns
// one 16-col tile with 2 M-subtiles x 6 gates = 12 accumulators.
// No LDS, no barriers: pure global-load + WMMA stream (fully pipelineable).
// ---------------------------------------------------------------------------
__global__ __launch_bounds__(256)
void k_gru(const unsigned short* __restrict__ out_bf,
           const unsigned short* __restrict__ h_bf,
           const unsigned short* __restrict__ Wih_bf,
           const float* __restrict__ WihLast,
           const unsigned short* __restrict__ Whh_bf,
           const float* __restrict__ bih, const float* __restrict__ bhh,
           const float* __restrict__ tension, const float* __restrict__ hiddens,
           float* __restrict__ new_h) {
  const int l     = blockIdx.x >> 8;          // 256 WGs per level
  const int rem   = blockIdx.x & 255;
  const int c0    = (rem >> 3) * 32;          // 32 cell tiles
  const int ncol0 = (rem & 7) * 128;          // 8 col tiles of 128
  const int w     = threadIdx.x >> 5;
  const int lane  = threadIdx.x & 31;
  const int colw  = ncol0 + w * 16;           // wave's h-column base

  v8f aI[3][2], aH[3][2];
  for (int g = 0; g < 3; ++g)
    for (int mt = 0; mt < 2; ++mt) { aI[g][mt] = (v8f){}; aH[g][mt] = (v8f){}; }

  const unsigned short* WihL  = Wih_bf + (size_t)l * G3 * OUT_D;
  const unsigned short* WhhL  = Whh_bf + (size_t)l * G3 * HID;
  const unsigned short* Aout  = out_bf + (size_t)(l * CPL + c0) * OUT_D;
  const unsigned short* Ahid  = h_bf   + (size_t)(l * CPL + c0) * HID;

  // gi = out @ Wih[:, :512]^T, K = 512
  for (int kt = 0; kt < OUT_D; kt += 32) {
    bfrag_t a0 = load_frag(Aout, 0,  OUT_D, kt);
    bfrag_t a1 = load_frag(Aout, 16, OUT_D, kt);
    for (int g = 0; g < 3; ++g) {
      bfrag_t b = load_frag(WihL, g * HID + colw, OUT_D, kt);
      aI[g][0] = wmma_bf16(a0, b, aI[g][0]);
      aI[g][1] = wmma_bf16(a1, b, aI[g][1]);
    }
  }

  // gh = h @ Whh^T, K = 1024
  for (int kt = 0; kt < HID; kt += 32) {
    bfrag_t a0 = load_frag(Ahid, 0,  HID, kt);
    bfrag_t a1 = load_frag(Ahid, 16, HID, kt);
    for (int g = 0; g < 3; ++g) {
      bfrag_t b = load_frag(WhhL, g * HID + colw, HID, kt);
      aH[g][0] = wmma_bf16(a0, b, aH[g][0]);
      aH[g][1] = wmma_bf16(a1, b, aH[g][1]);
    }
  }

  // ---- gates (torch order r,z,n) + rank-1 tension column + biases ----
  const int coln = lane & 15;
  const int hi8  = (lane >> 4) * 8;
  const int col  = colw + coln;
  const float wLr = WihLast[l * G3 + col];
  const float wLz = WihLast[l * G3 + HID + col];
  const float wLn = WihLast[l * G3 + 2 * HID + col];
  const float bir = bih[l * G3 + col];
  const float biz = bih[l * G3 + HID + col];
  const float bin = bih[l * G3 + 2 * HID + col];
  const float bhr = bhh[l * G3 + col];
  const float bhz = bhh[l * G3 + HID + col];
  const float bhn = bhh[l * G3 + 2 * HID + col];

  for (int mt = 0; mt < 2; ++mt)
    for (int i = 0; i < 8; ++i) {
      int m    = mt * 16 + hi8 + i;
      int cell = l * CPL + c0 + m;
      float tn  = tension[cell];
      float ir  = aI[0][mt][i] + tn * wLr + bir;
      float iz  = aI[1][mt][i] + tn * wLz + biz;
      float in_ = aI[2][mt][i] + tn * wLn + bin;
      float hr  = aH[0][mt][i] + bhr;
      float hz  = aH[1][mt][i] + bhz;
      float hn  = aH[2][mt][i] + bhn;
      float rg  = 1.0f / (1.0f + __expf(-(ir + hr)));
      float zg  = 1.0f / (1.0f + __expf(-(iz + hz)));
      float ng  = tanhf(in_ + rg * hn);
      float hv  = hiddens[(size_t)cell * HID + col];
      new_h[(size_t)cell * HID + col] = (1.0f - zg) * ng + zg * hv;
    }
}

// ---------------------------------------------------------------------------
// Small scalar tail kernels
// ---------------------------------------------------------------------------
__global__ void k_lmean(const float* __restrict__ new_h, float* __restrict__ lm) {
  int i = blockIdx.x * 256 + threadIdx.x;     // 4096 = [L][HID]
  if (i >= L_LV * HID) return;
  int l = i >> 10, h = i & 1023;
  float s = 0.0f;
  const float* p = new_h + (size_t)l * CPL * HID + h;
  for (int c = 0; c < CPL; ++c) s += p[(size_t)c * HID];
  lm[i] = s * (1.0f / CPL);
}

__global__ void k_pred1(const float* __restrict__ lm, const float* __restrict__ W1,
                        const float* __restrict__ b1, float* __restrict__ pz) {
  int i = blockIdx.x * 256 + threadIdx.x;     // 3*1024
  if (i >= (L_LV - 1) * HID) return;
  int l = i >> 10, m = i & 1023;
  const float* lv = lm + (l + 1) * HID;
  const float* w  = W1 + (size_t)l * HID * HID + m;
  float s = b1[i];
  for (int h = 0; h < HID; ++h) s += lv[h] * w[(size_t)h * HID];
  pz[i] = s > 0.0f ? s : 0.0f;
}

__global__ void k_pred2(const float* __restrict__ pz, const float* __restrict__ W2,
                        const float* __restrict__ b2, const float* __restrict__ lm,
                        float* __restrict__ pe) {
  int i = blockIdx.x * 256 + threadIdx.x;     // 3*1024
  if (i >= (L_LV - 1) * HID) return;
  int l = i >> 10, o = i & 1023;
  const float* z = pz + l * HID;
  const float* w = W2 + (size_t)l * HID * HID + o;
  float s = b2[i];
  for (int m = 0; m < HID; ++m) s += z[m] * w[(size_t)m * HID];
  pe[i] = (s - lm[l * HID + o]) * 0.05f;
}

// addterm[l][h] = 0.7225*delta + 0.2775*(lm+delta);  gop[h] = mean_l(lm+delta)
// block 0 also reduces avg_tension into d_out[512].
__global__ void k_delta(const float* __restrict__ lm, const float* __restrict__ pe,
                        const float* __restrict__ tension,
                        float* __restrict__ addterm, float* __restrict__ gop,
                        float* __restrict__ avg_out) {
  __shared__ float red[256];
  int h = blockIdx.x * 256 + threadIdx.x;     // 1024 threads total
  if (h < HID) {
    float g = 0.0f;
    for (int l = 0; l < L_LV; ++l) {
      float d = 0.0f;
      if (l < L_LV - 1) d += pe[l * HID + h];
      if (l > 0)        d -= 0.5f * pe[(l - 1) * HID + h];
      float ml = lm[l * HID + h] + d;
      addterm[l * HID + h] = 0.7225f * d + 0.2775f * ml;
      g += ml;
    }
    gop[h] = g * (1.0f / L_LV);
  }
  if (blockIdx.x == 0) {
    float s = 0.0f;
    for (int j = threadIdx.x; j < L_LV * CPL; j += 256) s += tension[j];
    red[threadIdx.x] = s;
    __syncthreads();
    for (int off = 128; off > 0; off >>= 1) {
      if (threadIdx.x < off) red[threadIdx.x] += red[threadIdx.x + off];
      __syncthreads();
    }
    if (threadIdx.x == 0) avg_out[0] = red[0] * (1.0f / (L_LV * CPL));
  }
}

// hh = 0.7225*new_h + addterm[l][h]; debate for c < C/4 when step > 5
__global__ void k_hh(const float* __restrict__ new_h, const float* __restrict__ addterm,
                     const float* __restrict__ gop, const int* __restrict__ step,
                     float* __restrict__ hh_out) {
  size_t idx = (size_t)blockIdx.x * 256 + threadIdx.x;
  if (idx >= (size_t)L_LV * CPL * HID) return;
  int h    = (int)(idx & 1023);
  int cell = (int)(idx >> 10);
  int l    = cell >> 10;
  int c    = cell & 1023;
  float v = 0.7225f * new_h[idx] + addterm[l * HID + h];
  if (*step > 5 && c < CPL / 4) v = 0.85f * v + 0.15f * gop[h];
  hh_out[idx] = v;
}

__global__ void k_combined(const float* __restrict__ lm, const float* __restrict__ pei_W,
                           const float* __restrict__ pei_b, float* __restrict__ out) {
  int o = blockIdx.x * 256 + threadIdx.x;     // 512
  if (o >= OUT_D) return;
  float s = pei_b[o];
  for (int j = 0; j < L_LV * HID; ++j) s += lm[j] * pei_W[(size_t)j * OUT_D + o];
  out[o] = s;
}

// ---------------------------------------------------------------------------
// Host launch
// ---------------------------------------------------------------------------
extern "C" void kernel_launch(void* const* d_in, const int* in_sizes, int n_in,
                              void* d_out, int out_size, void* d_ws, size_t ws_size,
                              hipStream_t stream) {
  const float* x        = (const float*)d_in[0];
  const float* hiddens  = (const float*)d_in[1];
  const float* ea_W1    = (const float*)d_in[2];
  const float* ea_b1    = (const float*)d_in[3];
  const float* ea_W2    = (const float*)d_in[4];
  const float* ea_b2    = (const float*)d_in[5];
  const float* eg_W1    = (const float*)d_in[6];
  const float* eg_b1    = (const float*)d_in[7];
  const float* eg_W2    = (const float*)d_in[8];
  const float* eg_b2    = (const float*)d_in[9];
  const float* gru_Wih  = (const float*)d_in[10];
  const float* gru_Whh  = (const float*)d_in[11];
  const float* gru_bih  = (const float*)d_in[12];
  const float* gru_bhh  = (const float*)d_in[13];
  const float* pred_W1  = (const float*)d_in[14];
  const float* pred_b1  = (const float*)d_in[15];
  const float* pred_W2  = (const float*)d_in[16];
  const float* pred_b2  = (const float*)d_in[17];
  const float* pei_W    = (const float*)d_in[18];
  const float* pei_b    = (const float*)d_in[19];
  const int*   step     = (const int*)d_in[20];
  float*       fout     = (float*)d_out;

  char*  ws  = (char*)d_ws;
  size_t off = 0;
  auto take = [&](size_t bytes) -> char* {
    char* p = ws + off;
    off = (off + bytes + 255) & ~(size_t)255;
    return p;
  };
  unsigned short* WA1t    = (unsigned short*)take((size_t)L_LV * MLP_D * COMB_D * 2);
  unsigned short* WG1t    = (unsigned short*)take((size_t)L_LV * MLP_D * COMB_D * 2);
  unsigned short* WA2t    = (unsigned short*)take((size_t)L_LV * OUT_D * MLP_D * 2);
  unsigned short* WG2t    = (unsigned short*)take((size_t)L_LV * OUT_D * MLP_D * 2);
  unsigned short* Wih_bf  = (unsigned short*)take((size_t)L_LV * G3 * OUT_D * 2);
  float*          WihLast = (float*)take((size_t)L_LV * G3 * 4);
  unsigned short* Whh_bf  = (unsigned short*)take((size_t)L_LV * G3 * HID * 2);
  unsigned short* x_bf    = (unsigned short*)take((size_t)IN_D * 2);
  unsigned short* h_bf    = (unsigned short*)take((size_t)L_LV * CPL * HID * 2);
  unsigned short* out_bf  = (unsigned short*)take((size_t)L_LV * CPL * OUT_D * 2);
  float*          tensn   = (float*)take((size_t)L_LV * CPL * 4);
  float*          new_h   = (float*)take((size_t)L_LV * CPL * HID * 4);
  float*          lm      = (float*)take((size_t)L_LV * HID * 4);
  float*          pz      = (float*)take((size_t)(L_LV - 1) * HID * 4);
  float*          pe      = (float*)take((size_t)(L_LV - 1) * HID * 4);
  float*          addterm = (float*)take((size_t)L_LV * HID * 4);
  float*          gop     = (float*)take((size_t)HID * 4);

  auto nblk = [](long long n) { return (int)((n + 255) / 256); };

  // conversions / transposes
  k_cvt_bf16<<<nblk(IN_D), 256, 0, stream>>>(x, x_bf, IN_D);
  k_cvt_bf16<<<nblk((long long)L_LV * CPL * HID), 256, 0, stream>>>(
      hiddens, h_bf, L_LV * CPL * HID);
  k_cvt_bf16<<<nblk((long long)L_LV * G3 * HID), 256, 0, stream>>>(
      gru_Whh, Whh_bf, L_LV * G3 * HID);
  k_transpose_W1<<<nblk((long long)L_LV * MLP_D * COMB_D), 256, 0, stream>>>(ea_W1, WA1t);
  k_transpose_W1<<<nblk((long long)L_LV * MLP_D * COMB_D), 256, 0, stream>>>(eg_W1, WG1t);
  k_transpose_W2<<<nblk((long long)L_LV * OUT_D * MLP_D), 256, 0, stream>>>(ea_W2, WA2t);
  k_transpose_W2<<<nblk((long long)L_LV * OUT_D * MLP_D), 256, 0, stream>>>(eg_W2, WG2t);
  k_cvt_wih<<<nblk((long long)L_LV * G3 * OUT_D), 256, 0, stream>>>(gru_Wih, Wih_bf, WihLast);

  // WMMA phases
  k_mlp<<<L_LV * (CPL / 64), 256, 0, stream>>>(x_bf, h_bf, WA1t, ea_b1, WA2t, ea_b2,
                                               WG1t, eg_b1, WG2t, eg_b2, out_bf, tensn);
  k_gru<<<L_LV * (CPL / 32) * (HID / 128), 256, 0, stream>>>(
      out_bf, h_bf, Wih_bf, WihLast, Whh_bf, gru_bih, gru_bhh, tensn, hiddens, new_h);

  // tail
  k_lmean<<<nblk(L_LV * HID), 256, 0, stream>>>(new_h, lm);
  k_pred1<<<nblk((L_LV - 1) * HID), 256, 0, stream>>>(lm, pred_W1, pred_b1, pz);
  k_pred2<<<nblk((L_LV - 1) * HID), 256, 0, stream>>>(pz, pred_W2, pred_b2, lm, pe);
  k_delta<<<nblk(HID), 256, 0, stream>>>(lm, pe, tensn, addterm, gop, fout + 512);
  k_hh<<<nblk((long long)L_LV * CPL * HID), 256, 0, stream>>>(new_h, addterm, gop,
                                                              step, fout + 513);
  k_combined<<<nblk(OUT_D), 256, 0, stream>>>(lm, pei_W, pei_b, fout);
}
